// DeepGCN_slic_V2_1_23845658427415
// MI455X (gfx1250) — matrix operand order repaired and checked by reading, not compiled, not run on hardware
//
#include <hip/hip_runtime.h>

// ---------------------------------------------------------------------------
// MI455X (gfx1250) implementation of Stem_slic_V2.compute_features.
// Heavy phase = batched GEMM [768x196]x[196x196] per batch, mapped onto
// v_wmma_f32_16x16x32_bf16 (wave32 WMMA). wcnt entries (ints <=16) are exact
// in bf16; conv_feat is rounded RNE to bf16; accumulation is f32.
// ---------------------------------------------------------------------------

typedef __attribute__((ext_vector_type(16))) __bf16        v16bf;
typedef __attribute__((ext_vector_type(8)))  float         v8f;
typedef __attribute__((ext_vector_type(4)))  unsigned int  v4u;

union Frag { v4u u[2]; v16bf v; };

__device__ __forceinline__ unsigned short f2bf(float f) {
  unsigned int u = __float_as_uint(f);
  u += 0x7FFFu + ((u >> 16) & 1u);   // round-to-nearest-even
  return (unsigned short)(u >> 16);
}
__device__ __forceinline__ float bf2f(unsigned short h) {
  return __uint_as_float(((unsigned int)h) << 16);
}

// problem constants
constexpr int NB   = 64;    // batch
constexpr int NC   = 768;   // channels (GEMM M-rows)
constexpr int HW   = 224;   // segmap size
constexpr int NR   = 14;    // window grid
constexpr int NSEG = 196;   // labels == windows
constexpr int KP   = 224;   // padded K (= pq) dimension: 7 x 32
constexpr int MP   = 208;   // padded label dimension: 13 x 16
constexpr int NT   = 13;    // N tiles of 16

// workspace layout (bytes)
constexpr size_t A_BYTES = (size_t)NB * NC * KP * 2;       // 22,020,096
constexpr size_t B_BYTES = (size_t)NB * MP * KP * 2;       //  5,963,776
// inv: NB * MP floats after the above

// ---------------------------------------------------------------------------
// Kernel 0: conv_feat f32 [B,768,196] -> bf16 [B,768,224] (zero K-pad)
// ---------------------------------------------------------------------------
__global__ void k_acvt(const float* __restrict__ feat,
                       unsigned int* __restrict__ aws /* packed 2xbf16 */) {
  size_t tid = (size_t)blockIdx.x * blockDim.x + threadIdx.x;
  size_t n2  = (size_t)NB * NC * KP / 2;
  if (tid >= n2) return;
  size_t i2  = tid * 2;
  size_t row = i2 / KP;
  int    k   = (int)(i2 % KP);
  float v0 = (k     < NSEG) ? feat[row * NSEG + k]     : 0.0f;
  float v1 = (k + 1 < NSEG) ? feat[row * NSEG + k + 1] : 0.0f;
  aws[tid] = (unsigned int)f2bf(v0) | ((unsigned int)f2bf(v1) << 16);
}

// ---------------------------------------------------------------------------
// Kernel 1: build wcnt, stored TRANSPOSED+padded as bf16 [B][m=208][pq=224].
// Grid: NB*16 blocks; p<14 computes window-row p, p in {14,15} zero-fills the
// pq pad columns. 224 threads = one cell each (14 windows x 16 cells).
// ---------------------------------------------------------------------------
__global__ void k_wcnt(const int* __restrict__ seg,
                       unsigned short* __restrict__ bws) {
  const int b = blockIdx.x >> 4;
  const int p = blockIdx.x & 15;
  __shared__ unsigned int cnt[NR * MP];          // [q][m], 11.6 KB
  const int tid = threadIdx.x;                   // 224 threads
  for (int i = tid; i < NR * MP; i += 224) cnt[i] = 0u;
  __syncthreads();

  if (p < NR) {
    const int q  = tid >> 4;
    const int c  = tid & 15;
    const int hc = 4 * p + (c >> 2);             // cell row in 56-grid
    const int wc = 4 * q + (c & 3);              // cell col in 56-grid
    const int* sb = seg + (size_t)b * HW * HW;
    const int r0 = 4 * hc + 1, c0 = 4 * wc + 1;  // corner pixels (1::4 / 2::4)
    int l0 = sb[(size_t)r0 * HW + c0];
    int l1 = sb[(size_t)r0 * HW + c0 + 1];
    int l2 = sb[(size_t)(r0 + 1) * HW + c0];
    int l3 = sb[(size_t)(r0 + 1) * HW + c0 + 1];
    // "any corner matches" => count each distinct label once per cell
    atomicAdd(&cnt[q * MP + l0], 1u);
    if (l1 != l0)                           atomicAdd(&cnt[q * MP + l1], 1u);
    if (l2 != l0 && l2 != l1)               atomicAdd(&cnt[q * MP + l2], 1u);
    if (l3 != l0 && l3 != l1 && l3 != l2)   atomicAdd(&cnt[q * MP + l3], 1u);
  }
  __syncthreads();

  // emit transposed: bws[b][m][p*14+q]
  for (int i = tid; i < MP * NR; i += 224) {
    const int m = i / NR, q = i - m * NR;
    unsigned int v = (p < NR && m < NSEG) ? cnt[q * MP + m] : 0u;
    bws[((size_t)b * MP + m) * KP + p * NR + q] = f2bf((float)v);
  }
}

// ---------------------------------------------------------------------------
// Kernel 2: per-label reciprocal counts (0 where count==0 -> nan_to_num).
// ---------------------------------------------------------------------------
__global__ void k_inv(const unsigned short* __restrict__ bws,
                      float* __restrict__ inv) {
  const int b = blockIdx.x;
  const int m = threadIdx.x;
  if (m >= MP) return;
  float r = 0.0f;
  if (m < NSEG) {
    const unsigned short* row = bws + ((size_t)b * MP + m) * KP;
    float s = 0.0f;                               // exact integer sum
    for (int k = 0; k < NSEG; ++k) s += bf2f(row[k]);
    r = (s > 0.0f) ? 1.0f / s : 0.0f;
  }
  inv[b * MP + m] = r;
}

// ---------------------------------------------------------------------------
// Kernel 3: WMMA GEMM  out[b] = feat_bf16[b] (768x224) x wcnt[b] (224x208),
// column-scaled by inv. 256 threads = 8 wave32; wave owns a 16-row tile and
// all 13 N-tiles (acc stays resident), K loop = 7 steps of 32.
// ---------------------------------------------------------------------------
__global__ void __launch_bounds__(256)
k_gemm(const unsigned short* __restrict__ aws,
       const unsigned short* __restrict__ bws,
       const float* __restrict__ inv,
       float* __restrict__ out) {
  const int b      = blockIdx.x / 6;
  const int stripe = blockIdx.x % 6;             // 6 stripes x 128 rows
  const int wave   = threadIdx.x >> 5;
  const int lane   = threadIdx.x & 31;
  const int half   = lane >> 4;                  // lane group (ISA layout)
  const int lid    = lane & 15;

  // A fragment source row for this lane (16-bit A 16x32 layout, ISA 7.12.2):
  // lanes 0-15 hold K {0..7,16..23}, lanes 16-31 hold K {8..15,24..31}.
  const int arow = stripe * 128 + wave * 16 + lid;
  const unsigned short* aptr = aws + ((size_t)b * NC + arow) * KP + half * 8;
  // B fragment: lane = column N, same K order as A (K-major storage).
  const unsigned short* bptr = bws + (size_t)b * MP * KP + half * 8;

  v8f acc[NT] = {};

#pragma unroll
  for (int k0 = 0; k0 < KP; k0 += 32) {
    Frag a;
    const v4u* pa = (const v4u*)(aptr + k0);
    a.u[0] = pa[0];                              // K = k0 + half*8 + 0..7
    a.u[1] = pa[2];                              // K = k0 + 16 + half*8 + 0..7
#pragma unroll
    for (int nt = 0; nt < NT; ++nt) {
      Frag bb;
      const v4u* pb = (const v4u*)(bptr + (size_t)(nt * 16 + lid) * KP + k0);
      bb.u[0] = pb[0];                           // K = k0 + half*8 + 0..7
      bb.u[1] = pb[2];                           // K = k0 + 16 + half*8 + 0..7
      acc[nt] = __builtin_amdgcn_wmma_f32_16x16x32_bf16(
          false, a.v, false, bb.v, (short)0, acc[nt], false, false);
    }
  }

  // D layout: lanes 0-15 VGPR r -> M=r ; lanes 16-31 -> M=8+r. Col = N lane.
  const int rbase = stripe * 128 + wave * 16 + half * 8;
#pragma unroll
  for (int nt = 0; nt < NT; ++nt) {
    const int col = nt * 16 + lid;
    if (col < NSEG) {
      const float s = inv[b * MP + col];
#pragma unroll
      for (int r = 0; r < 8; ++r)
        out[((size_t)b * NC + rbase + r) * NSEG + col] = acc[nt][r] * s;
    }
  }
}

// ---------------------------------------------------------------------------
extern "C" void kernel_launch(void* const* d_in, const int* in_sizes, int n_in,
                              void* d_out, int out_size, void* d_ws, size_t ws_size,
                              hipStream_t stream) {
  (void)in_sizes; (void)n_in; (void)out_size; (void)ws_size;
  const int*   seg  = (const int*)d_in[0];     // [64,224,224] int32
  const float* feat = (const float*)d_in[1];   // [64,768,14,14] f32
  float*       out  = (float*)d_out;           // [64,768,14,14] f32

  unsigned short* aws = (unsigned short*)d_ws;
  unsigned short* bws = (unsigned short*)((char*)d_ws + A_BYTES);
  float*          inv = (float*)((char*)d_ws + A_BYTES + B_BYTES);

  const int n2blocks = (int)(((size_t)NB * NC * KP / 2) / 256);  // 21504
  k_acvt<<<n2blocks, 256, 0, stream>>>(feat, (unsigned int*)aws);
  k_wcnt<<<NB * 16, 224, 0, stream>>>(seg, bws);
  k_inv <<<NB, 224, 0, stream>>>(bws, inv);
  k_gemm<<<NB * 6, 256, 0, stream>>>(aws, bws, inv, out);
}
